// S4Layer_2439541424623
// MI455X (gfx1250) — matrix operand order repaired and checked by reading, not compiled, not run on hardware
//
#include <hip/hip_runtime.h>
#include <hip/hip_bf16.h>

// Problem constants (match reference: B_SZ=8, H=256, L_MAX=4096, N=64)
#define Hh 256
#define Ll 4096
#define Nn 64
#define Bb 8
#define HL (Hh * Ll)
#define UTOT (Bb * Hh * Ll)

typedef __attribute__((ext_vector_type(16))) __bf16 v16bf;
typedef __attribute__((ext_vector_type(8)))  float  v8f;

__device__ __forceinline__ unsigned int bf16bits(float f) {
  __bf16 h = (__bf16)f;
  unsigned short s;
  __builtin_memcpy(&s, &h, 2);
  return (unsigned int)s;
}

// ---------------------------------------------------------------------------
// Stage 0: pre-convert u (f32) -> bf16 bit image (one pass, bandwidth bound).
// Each thread packs 8 contiguous floats into one uint4 store.
// ---------------------------------------------------------------------------
__global__ void s4_u2bf(const float* __restrict__ u, uint4* __restrict__ ubf) {
  const int i = blockIdx.x * 256 + threadIdx.x;      // group of 8 elements
  const float4* __restrict__ s = (const float4*)(u + (size_t)i * 8);
  const float4 x0 = s[0], x1 = s[1];
  uint4 o;
  o.x = bf16bits(x0.x) | (bf16bits(x0.y) << 16);
  o.y = bf16bits(x0.z) | (bf16bits(x0.w) << 16);
  o.z = bf16bits(x1.x) | (bf16bits(x1.y) << 16);
  o.w = bf16bits(x1.z) | (bf16bits(x1.w) << 16);
  ubf[i] = o;
}

// ---------------------------------------------------------------------------
// Stage 1: atRoots(h,l) — DPLR generating function at the L roots of unity.
// ---------------------------------------------------------------------------
__global__ void s4_atroots(const float* __restrict__ Lre, const float* __restrict__ Lim,
                           const float* __restrict__ Pm,  const float* __restrict__ Bm,
                           const float* __restrict__ Cm,  const float* __restrict__ lstep,
                           float* __restrict__ Ar, float* __restrict__ Ai) {
  __shared__ float sLr[Nn], sLi[Nn];
  __shared__ float s00r[Nn], s00i[Nn], s01r[Nn], s01i[Nn];
  __shared__ float s10r[Nn], s10i[Nn], s11r[Nn], s11i[Nn];

  const int h = blockIdx.x >> 4;                 // 16 chunks of 256 l's per head
  const int l = ((blockIdx.x & 15) << 8) + threadIdx.x;

  if (threadIdx.x < Nn) {
    const int n   = threadIdx.x;
    const int idx = h * Nn + n;
    const float lr = fminf(Lre[idx], -1e-4f), li = Lim[idx];
    const float pr = Pm[2*idx], pi = Pm[2*idx+1];
    const float br = Bm[2*idx], bi = Bm[2*idx+1];
    const float cr = Cm[2*idx], ci = Cm[2*idx+1];
    sLr[n] = lr; sLi[n] = li;
    s00r[n] = cr*br + ci*bi;  s00i[n] = cr*bi - ci*br;   // conj(C)*B
    s01r[n] = cr*pr + ci*pi;  s01i[n] = cr*pi - ci*pr;   // conj(C)*P
    s10r[n] = pr*br + pi*bi;  s10i[n] = pr*bi - pi*br;   // conj(P)*B
    s11r[n] = pr*pr + pi*pi;  s11i[n] = 0.0f;            // conj(P)*P
  }
  __syncthreads();

  const float step = expf(lstep[h]);
  float sn, cs;
  __sincosf(6.283185307179586f * (float)l * (1.0f / (float)Ll), &sn, &cs);
  // Omega = exp(-i*theta) = (cs, -sn)
  const float dr = 1.0f + cs, di = -sn;          // 1 + Omega
  const float nr = 1.0f - cs, ni =  sn;          // 1 - Omega
  const float inv = 1.0f / (dr*dr + di*di);
  const float sc  = 2.0f / step;
  const float gr = sc * (nr*dr + ni*di) * inv;   // g = (2/step)(1-O)/(1+O)
  const float gi = sc * (ni*dr - nr*di) * inv;
  const float c_r =  2.0f * dr * inv;            // c = 2/(1+Omega)
  const float c_i = -2.0f * di * inv;

  float k00r=0.f,k00i=0.f,k01r=0.f,k01i=0.f,k10r=0.f,k10i=0.f,k11r=0.f,k11i=0.f;
  #pragma unroll 8
  for (int n = 0; n < Nn; ++n) {
    const float er = gr - sLr[n], ei = gi - sLi[n];
    const float iv = 1.0f / (er*er + ei*ei);
    const float rr =  er * iv, ri = -ei * iv;    // 1/(g - Lambda)
    k00r += s00r[n]*rr - s00i[n]*ri;  k00i += s00r[n]*ri + s00i[n]*rr;
    k01r += s01r[n]*rr - s01i[n]*ri;  k01i += s01r[n]*ri + s01i[n]*rr;
    k10r += s10r[n]*rr - s10i[n]*ri;  k10i += s10r[n]*ri + s10i[n]*rr;
    k11r += s11r[n]*rr - s11i[n]*ri;  k11i += s11r[n]*ri + s11i[n]*rr;
  }
  // at = c * (k00 - k01 * k10 / (1 + k11))
  const float d1r = 1.0f + k11r, d1i = k11i;
  const float iv1 = 1.0f / (d1r*d1r + d1i*d1i);
  const float qr = k01r*k10r - k01i*k10i, qi = k01r*k10i + k01i*k10r;
  const float tr = (qr*d1r + qi*d1i) * iv1, ti = (qi*d1r - qr*d1i) * iv1;
  const float mr = k00r - tr, mi = k00i - ti;
  Ar[h*Ll + l] = c_r*mr - c_i*mi;
  Ai[h*Ll + l] = c_r*mi + c_i*mr;
}

// ---------------------------------------------------------------------------
// Stage 2: K[h,t] = Re(ifft(atRoots[h,:]))  (direct inverse DFT; exact integer
// phase:  angle = 2*pi*((l*t) mod L)/L)
// ---------------------------------------------------------------------------
__global__ void s4_idft(const float* __restrict__ Ar, const float* __restrict__ Ai,
                        float* __restrict__ K) {
  __shared__ float sr[Ll];
  __shared__ float si[Ll];
  const int h = blockIdx.x >> 4;
  const int t = ((blockIdx.x & 15) << 8) + threadIdx.x;
  for (int i = threadIdx.x; i < Ll; i += 256) {
    sr[i] = Ar[h*Ll + i];
    si[i] = Ai[h*Ll + i];
  }
  __syncthreads();
  const float w = 6.283185307179586f / (float)Ll;
  float acc = 0.0f;
  for (int l2 = 0; l2 < Ll; ++l2) {
    const int p = (l2 * t) & (Ll - 1);
    float sn, cs;
    __sincosf(w * (float)p, &sn, &cs);
    acc += sr[l2]*cs - si[l2]*sn;   // Re( A[l] * e^{+2pi i l t / L} )
  }
  K[h*Ll + t] = acc * (1.0f / (float)Ll);
}

// ---------------------------------------------------------------------------
// Stage 3: causal conv y = (u * K)[:L] + D*u as per-head GEMM on WMMA:
//   D[m,n] = sum_k A[m,k] * B[k,n]
//   A[m,k] = ubf[batch m, j0+k]          (pre-converted bf16; 2x b128 loads)
//   B[k,n] = Kext[i0 + n - j0 - k]       (Toeplitz, from reversed zero-padded
//                                         LDS image; branch-free aligned b32)
// Output tile: row m = batch (m<8 valid), column n = output offset.
// ---------------------------------------------------------------------------
#define RN 2080   // uint32 words: covers reversed K plus 64 zero-pad entries

__global__ void s4_conv(const float* __restrict__ Kf,
                        const unsigned short* __restrict__ ubf,
                        const float* __restrict__ u,
                        const float* __restrict__ Dv, float* __restrict__ y) {
  // R[j] = Kext[4095 - j], j in [0, 4160); Kext[x<0] = 0.
  // R0[i] packs (R[2i], R[2i+1]); R1[i] packs (R[2i+1], R[2i+2]).
  __shared__ unsigned int R0[RN];
  __shared__ unsigned int R1[RN];

  const int h   = blockIdx.x >> 5;               // 32 row-groups (of 8 waves) per head
  const int rg  = blockIdx.x & 31;
  const int tid = threadIdx.x;
  const float* __restrict__ Kh = Kf + (size_t)h * Ll;

  for (int i = tid; i < RN; i += 256) {
    const int j2 = 2 * i;
    const float f0 = (4095 - j2 >= 0) ? Kh[4095 - j2] : 0.f;  // R[2i]
    const float f1 = (4094 - j2 >= 0) ? Kh[4094 - j2] : 0.f;  // R[2i+1]
    const float f2 = (4093 - j2 >= 0) ? Kh[4093 - j2] : 0.f;  // R[2i+2]
    const unsigned int b0 = bf16bits(f0), b1 = bf16bits(f1), b2 = bf16bits(f2);
    R0[i] = b0 | (b1 << 16);
    R1[i] = b1 | (b2 << 16);
  }
  __syncthreads();

  const int lane = tid & 31;
  const int wave = tid >> 5;
  const int i0   = (((rg << 3) + wave) << 4);    // output column-block start (16)
  const int hs   = lane >> 4;                    // half-wave select
  const int ln   = lane & 15;                    // A: batch row m; B/D: column n

  // A operand: bf16 u rows (batch = ln; clamp >=8 to row 0, rows 8..15 dead).
  const int nc = (ln < Bb) ? ln : 0;
  const unsigned short* __restrict__ ucol =
      ubf + ((size_t)nc * Hh + h) * (size_t)Ll;

  v8f acc = {0.f,0.f,0.f,0.f,0.f,0.f,0.f,0.f};

  const int jmax = i0 + 15;
  for (int j0 = 0; j0 <= jmax; j0 += 32) {
    // ---- A-tile: element e -> k = hs*16 + e; A[m,k]=u[m, j0+k].
    // Two aligned runs of 8 consecutive bf16 -> 2x 16-byte loads, no VALU.
    union { v16bf v; uint4 q[2]; } au;
    au.q[0] = *(const uint4*)(ucol + j0 + (hs << 4));
    au.q[1] = *(const uint4*)(ucol + j0 + 8 + (hs << 4));

    // ---- B-tile: element e -> k = (lane>>4)*16 + e; B[k,n]=Kext[i0+n-j0-k].
    // Reversed index j = 4095 - (i0+n-j0-k) ascends with e: 16 consecutive
    // bf16 -> 8 aligned uint32 from R0 (even start) or R1 (odd start).
    const int bidx = 4095 - i0 - ln + j0 + (hs << 4);
    const unsigned int* __restrict__ src =
        ((bidx & 1) ? R1 : R0) + (bidx >> 1);
    union { v16bf v; unsigned int w[8]; } bu;
    #pragma unroll
    for (int q = 0; q < 8; ++q) bu.w[q] = src[q];

    acc = __builtin_amdgcn_wmma_f32_16x16x32_bf16(
        /*neg_a=*/false, au.v, /*neg_b=*/false, bu.v,
        /*c_mod=*/(short)0, acc, /*reuse_a=*/false, /*reuse_b=*/false);
  }

  // ---- epilogue: C/D 16x16 f32 layout: lanes 0-15 -> N=lane, VGPR v -> M=v.
  // M = batch (0..7 valid), N = output offset.  Lanes 16-31 hold dead rows.
  if (hs == 0) {
    const float Dh = Dv[h];
    const int i = i0 + ln;
    #pragma unroll
    for (int v = 0; v < 8; ++v) {
      const float* __restrict__ ur = u + ((size_t)v * Hh + h) * (size_t)Ll;
      float* __restrict__ yr = y + ((size_t)v * Hh + h) * (size_t)Ll;
      yr[i] = acc[v] + Dh * ur[i];
    }
  }
}

// ---------------------------------------------------------------------------
// kernel_launch: inputs in setup_inputs() order:
//   0: u (8*256*4096)  1: Lambda_re (256*64)  2: Lambda_im  3: P (256*64*2)
//   4: B (256*64*2)    5: C (256*64*2)        6: log_step (256)  7: D (256)
// Workspace: Ar | Ai | K (3*H*L f32 = 12 MB) | ubf (B*H*L bf16 = 16 MB).
// ---------------------------------------------------------------------------
extern "C" void kernel_launch(void* const* d_in, const int* in_sizes, int n_in,
                              void* d_out, int out_size, void* d_ws, size_t ws_size,
                              hipStream_t stream) {
  const float* u   = (const float*)d_in[0];
  const float* Lre = (const float*)d_in[1];
  const float* Lim = (const float*)d_in[2];
  const float* Pm  = (const float*)d_in[3];
  const float* Bm  = (const float*)d_in[4];
  const float* Cm  = (const float*)d_in[5];
  const float* ls  = (const float*)d_in[6];
  const float* Dv  = (const float*)d_in[7];

  float* w  = (float*)d_ws;
  float* Ar = w;
  float* Ai = w + (size_t)HL;
  float* K  = w + (size_t)2 * HL;
  unsigned short* ubf = (unsigned short*)(w + (size_t)3 * HL);

  // Stage 0: u -> bf16 image — grid = UTOT / (8*256)
  s4_u2bf<<<UTOT / (8 * 256), 256, 0, stream>>>(u, (uint4*)ubf);
  // Stage 1: atRoots — grid = H * (L/256)
  s4_atroots<<<Hh * (Ll / 256), 256, 0, stream>>>(Lre, Lim, Pm, Bm, Cm, ls, Ar, Ai);
  // Stage 2: inverse DFT -> K — grid = H * (L/256)
  s4_idft<<<Hh * (Ll / 256), 256, 0, stream>>>(Ar, Ai, K);
  // Stage 3: WMMA Toeplitz causal conv + skip — grid = H * 32 (8 waves/block)
  s4_conv<<<Hh * 32, 256, 0, stream>>>(K, ubf, u, Dv, (float*)d_out);
}
